// Sub_MGAI_7902739825287
// MI455X (gfx1250) — matrix operand order repaired and checked by reading, not compiled, not run on hardware
//
#include <hip/hip_runtime.h>

// ---------------- problem dimensions ----------------
constexpr int B  = 16;
constexpr int C  = 256;
constexpr int N  = 4096;          // H*W
constexpr int K  = 8;             // clusters
constexpr int CN = C * N;
constexpr int KN = K * N;
constexpr int KC = K * C;
constexpr float EPS_BN = 1e-5f;

// ---------------- WMMA vector types (CDNA5, wave32) ----------------
typedef __attribute__((ext_vector_type(16))) __bf16 v16bf;
typedef __attribute__((ext_vector_type(8)))  float  v8f;

__device__ __forceinline__ float sigmoidf(float x) { return 1.f / (1.f + __expf(-x)); }
__device__ __forceinline__ float4 ld4(const float* __restrict__ p) { return *(const float4*)p; }

// 16 contiguous floats -> register array
__device__ __forceinline__ void ld16(const float* __restrict__ p, float f[16]) {
    float4 a = ld4(p), b = ld4(p + 4), c = ld4(p + 8), d = ld4(p + 12);
    f[0]=a.x; f[1]=a.y; f[2]=a.z; f[3]=a.w;  f[4]=b.x; f[5]=b.y; f[6]=b.z; f[7]=b.w;
    f[8]=c.x; f[9]=c.y; f[10]=c.z; f[11]=c.w; f[12]=d.x; f[13]=d.y; f[14]=d.z; f[15]=d.w;
}
// A-fragment pattern: runs at +0..7 and +16..23 (16-bit 16x32 A layout after hi*8 shift)
__device__ __forceinline__ void ld16_gap(const float* __restrict__ p, float f[16]) {
    float4 a = ld4(p), b = ld4(p + 4), c = ld4(p + 16), d = ld4(p + 20);
    f[0]=a.x; f[1]=a.y; f[2]=a.z; f[3]=a.w;  f[4]=b.x; f[5]=b.y; f[6]=b.z; f[7]=b.w;
    f[8]=c.x; f[9]=c.y; f[10]=c.z; f[11]=c.w; f[12]=d.x; f[13]=d.y; f[14]=d.z; f[15]=d.w;
}
__device__ __forceinline__ float4 f4fma(float s, float4 v, float4 a) {
    a.x += s * v.x; a.y += s * v.y; a.z += s * v.z; a.w += s * v.w; return a;
}

// ============================================================
// K1: gates (4 pixels/thread, b128 loads)
// ============================================================
__global__ void k_gate(const float* __restrict__ edger, const float* __restrict__ edget,
                       const float* __restrict__ pw, const float* __restrict__ pb,
                       float* __restrict__ gate_r, float* __restrict__ gate_t) {
    __shared__ float w[C];
    int tid = threadIdx.x;
    w[tid] = pw[tid];
    __syncthreads();
    int idx = blockIdx.x * 256 + tid;       // quad id, 0 .. B*N/4-1
    int b = idx >> 10;                      // N/4 = 1024 quads per batch
    int n = (idx & 1023) * 4;
    const float* er = edger + (size_t)b * CN + n;
    const float* et = edget + (size_t)b * CN + n;
    float4 ar = {0.f, 0.f, 0.f, 0.f}, at = ar;
    for (int c = 0; c < C; ++c) {
        ar = f4fma(w[c], ld4(er + (size_t)c * N), ar);
        at = f4fma(w[c], ld4(et + (size_t)c * N), at);
    }
    float bias = pb[0];
    float4 gr = { sigmoidf(ar.x + bias), sigmoidf(ar.y + bias), sigmoidf(ar.z + bias), sigmoidf(ar.w + bias) };
    float4 gt = { sigmoidf(at.x + bias), sigmoidf(at.y + bias), sigmoidf(at.z + bias), sigmoidf(at.w + bias) };
    *(float4*)(gate_r + (size_t)b * N + n) = gr;
    *(float4*)(gate_t + (size_t)b * N + n) = gt;
}

// ============================================================
// K2: tables (transposed [c][k] layout for vector LDS reads)
// ============================================================
__global__ void k_tables(const float* __restrict__ anchor, const float* __restrict__ sigma_p,
                         float* __restrict__ tabA, float* __restrict__ tabB, float* __restrict__ cst) {
    int k = blockIdx.x, c = threadIdx.x;
    float sg  = sigmoidf(sigma_p[k * C + c]);
    float inv = 1.f / (sg * sg);
    float a   = anchor[k * C + c];
    tabA[c * K + k] = inv;
    tabB[c * K + k] = -2.f * a * inv;
    __shared__ float red[256];
    red[c] = a * a * inv;
    __syncthreads();
    for (int s = 128; s > 0; s >>= 1) { if (c < s) red[c] += red[c + s]; __syncthreads(); }
    if (c == 0) cst[k] = red[0];
}

__global__ void k_zero(float* __restrict__ p, int n) {
    int i = blockIdx.x * 256 + threadIdx.x;
    if (i < n) p[i] = 0.f;
}

// ============================================================
// K3: soft assignment (4 pixels/thread), moments s and G via LDS atomics
// ============================================================
__global__ void k_assign(const float* __restrict__ xin, const float* __restrict__ gate,
                         const float* __restrict__ tabA, const float* __restrict__ tabB,
                         const float* __restrict__ cst,
                         float* __restrict__ assign, float* __restrict__ sred, float* __restrict__ Gred) {
    __shared__ float shA[KC], shB[KC];
    __shared__ float s_sh[K], G_sh[K * K];
    int tid = threadIdx.x;
    for (int i = tid; i < KC; i += 256) { shA[i] = tabA[i]; shB[i] = tabB[i]; }
    if (tid < K)     s_sh[tid] = 0.f;
    if (tid < K * K) G_sh[tid] = 0.f;
    __syncthreads();

    int idx = blockIdx.x * 256 + tid;       // quad id
    int b = idx >> 10;
    int n = (idx & 1023) * 4;
    float4 g4 = ld4(gate + (size_t)b * N + n);
    const float* xp = xin + (size_t)b * CN + n;

    float q[K * 4];
    #pragma unroll
    for (int k = 0; k < K; ++k) {
        float cv = cst[k];
        #pragma unroll
        for (int j = 0; j < 4; ++j) q[k * 4 + j] = cv;
    }
    for (int c = 0; c < C; ++c) {
        float4 xv = ld4(xp + (size_t)c * N);
        float xc[4]  = { g4.x * xv.x, g4.y * xv.y, g4.z * xv.z, g4.w * xv.w };
        float x2c[4] = { xc[0]*xc[0], xc[1]*xc[1], xc[2]*xc[2], xc[3]*xc[3] };
        float4 A0 = *(const float4*)(shA + c * K), A1 = *(const float4*)(shA + c * K + 4);
        float4 B0 = *(const float4*)(shB + c * K), B1 = *(const float4*)(shB + c * K + 4);
        float a4[8] = {A0.x,A0.y,A0.z,A0.w,A1.x,A1.y,A1.z,A1.w};
        float b4[8] = {B0.x,B0.y,B0.z,B0.w,B1.x,B1.y,B1.z,B1.w};
        #pragma unroll
        for (int k = 0; k < K; ++k) {
            #pragma unroll
            for (int j = 0; j < 4; ++j) q[k * 4 + j] += x2c[j] * a4[k] + xc[j] * b4[k];
        }
    }
    #pragma unroll
    for (int j = 0; j < 4; ++j) {
        float mx = -1e30f;
        #pragma unroll
        for (int k = 0; k < K; ++k) { float l = -0.5f * q[k * 4 + j]; q[k * 4 + j] = l; mx = fmaxf(mx, l); }
        float sm = 0.f;
        #pragma unroll
        for (int k = 0; k < K; ++k) { float e = __expf(q[k * 4 + j] - mx); q[k * 4 + j] = e; sm += e; }
        float inv = 1.f / sm;
        #pragma unroll
        for (int k = 0; k < K; ++k) q[k * 4 + j] *= inv;
    }
    #pragma unroll
    for (int k = 0; k < K; ++k) {
        float4 pv = { q[k*4+0], q[k*4+1], q[k*4+2], q[k*4+3] };
        *(float4*)(assign + (size_t)b * KN + (size_t)k * N + n) = pv;
    }
    #pragma unroll
    for (int k = 0; k < K; ++k)
        atomicAdd(&s_sh[k], q[k*4+0] + q[k*4+1] + q[k*4+2] + q[k*4+3]);
    #pragma unroll
    for (int k = 0; k < K; ++k) {
        #pragma unroll
        for (int l = 0; l < K; ++l) {
            float gs = q[k*4+0]*q[l*4+0] + q[k*4+1]*q[l*4+1] + q[k*4+2]*q[l*4+2] + q[k*4+3]*q[l*4+3];
            atomicAdd(&G_sh[k * K + l], gs);
        }
    }
    __syncthreads();
    if (tid < K)     atomicAdd(&sred[b * K + tid], s_sh[tid]);
    if (tid < K * K) atomicAdd(&Gred[b * K * K + tid], G_sh[tid]);
}

// ============================================================
// K4 (WMMA): wx = assign @ xf. Branch-free b128 fragment loads.
//   grid = B*4 (b, N-segment), block = 256 (8 waves x 2 col-tiles each).
//   Partials over 4 segments -> wxp[seg][b][K][C], summed in k_nodes.
// ============================================================
__global__ void k_wx(const float* __restrict__ xin, const float* __restrict__ gate,
                     const float* __restrict__ assign, float* __restrict__ wxp) {
    int b    = blockIdx.x >> 2;
    int seg  = blockIdx.x & 3;
    int w    = threadIdx.x >> 5;      // tile pair 0..7
    int lane = threadIdx.x & 31;
    int hi   = lane >> 4;
    int m    = lane & 15;
    int m8   = m & 7;
    float rowmask = (m < K) ? 1.f : 0.f;   // zero-pad rows 8..15 (v_cndmask, no EXEC churn)
    int c0 = w * 32 + (lane & 15);

    const float* arow = assign + (size_t)b * KN + (size_t)m8 * N;
    const float* x0   = xin + (size_t)b * CN + (size_t)c0 * N;
    const float* x1   = x0 + (size_t)16 * N;
    const float* gb   = gate + (size_t)b * N;

    v8f acc0 = {}, acc1 = {};
    for (int kb = seg * 32; kb < seg * 32 + 32; ++kb) {
        int nbase = kb * 32;
        float af[16]; ld16(arow + nbase + hi * 16, af);
        v16bf av;
        #pragma unroll
        for (int e = 0; e < 16; ++e) av[e] = (__bf16)(af[e] * rowmask);
        int nb2 = nbase + hi * 16;
        float gf[16];  ld16(gb + nb2, gf);
        float xf0[16]; ld16(x0 + nb2, xf0);
        float xf1[16]; ld16(x1 + nb2, xf1);
        v16bf bv0, bv1;
        #pragma unroll
        for (int e = 0; e < 16; ++e) {
            bv0[e] = (__bf16)(gf[e] * xf0[e]);
            bv1[e] = (__bf16)(gf[e] * xf1[e]);
        }
        acc0 = __builtin_amdgcn_wmma_f32_16x16x32_bf16(false, av, false, bv0, (short)0, acc0, false, false);
        acc1 = __builtin_amdgcn_wmma_f32_16x16x32_bf16(false, av, false, bv1, (short)0, acc1, false, false);
    }
    if (lane < 16) {
        float* dst = wxp + ((size_t)seg * B + b) * KC;
        #pragma unroll
        for (int r = 0; r < K; ++r) {
            dst[r * C + w * 32 + lane]      = acc0[r];
            dst[r * C + w * 32 + 16 + lane] = acc1[r];
        }
    }
}

// ============================================================
// K5: sum wx partials, VLAD residual + row L2 + full L2 -> graph
// ============================================================
__global__ void k_nodes(const float* __restrict__ wxp, const float* __restrict__ sred,
                        const float* __restrict__ anchor, const float* __restrict__ sigma_p,
                        float* __restrict__ graph) {
    int b = blockIdx.x, c = threadIdx.x;
    __shared__ float rn[K];
    __shared__ float sv[K];
    __shared__ float tot;
    if (c < K) { rn[c] = 0.f; sv[c] = sred[b * K + c]; }
    if (c == 0) tot = 0.f;
    __syncthreads();
    float val[K];
    #pragma unroll
    for (int k = 0; k < K; ++k) {
        size_t off = (size_t)k * C + c;
        float wv = wxp[((size_t)0 * B + b) * KC + off] + wxp[((size_t)1 * B + b) * KC + off]
                 + wxp[((size_t)2 * B + b) * KC + off] + wxp[((size_t)3 * B + b) * KC + off];
        float sg = sigmoidf(sigma_p[k * C + c]);
        float s  = sv[k];
        float v  = (wv - s * anchor[k * C + c]) / sg / (s + 1e-9f);
        val[k] = v;
        atomicAdd(&rn[k], v * v);
    }
    __syncthreads();
    if (c < K) rn[c] = 1.f / fmaxf(sqrtf(rn[c]), 1e-12f);
    __syncthreads();
    float loc = 0.f;
    #pragma unroll
    for (int k = 0; k < K; ++k) { val[k] *= rn[k]; loc += val[k] * val[k]; }
    atomicAdd(&tot, loc);
    __syncthreads();
    float ti = 1.f / fmaxf(sqrtf(tot), 1e-12f);
    #pragma unroll
    for (int k = 0; k < K; ++k) graph[(size_t)b * KC + k * C + c] = val[k] * ti;
}

// ============================================================
// K6a (WMMA): graph conv y[b] = W @ g[b]  ([256x256] @ [256x8->16 padded]).
//   grid = B, block = 512 (16 waves = 16 o-tiles); g staged zero-padded in LDS as bf16.
// ============================================================
__global__ void __launch_bounds__(512) k_ggemm(const float* __restrict__ g,
                                               const float* __restrict__ Wm,
                                               float* __restrict__ y) {
    __shared__ __bf16 shg[C * 16];
    int b = blockIdx.x;
    int tid = threadIdx.x;
    for (int idx = tid; idx < C * 16; idx += 512) {
        int i = idx >> 4, c = idx & 15;
        float v = (c < K) ? g[(size_t)b * KC + i * K + c] : 0.f;
        shg[idx] = (__bf16)v;
    }
    __syncthreads();
    int w = tid >> 5, lane = tid & 31;
    int hi = lane >> 4, m = lane & 15, col = lane & 15;
    const float* wr = Wm + (size_t)(w * 16 + m) * C;
    v8f acc = {};
    for (int kb = 0; kb < 8; ++kb) {
        int nbase = kb * 32;
        float af[16]; ld16(wr + nbase + hi * 16, af);
        v16bf av;
        #pragma unroll
        for (int e = 0; e < 16; ++e) av[e] = (__bf16)af[e];
        int nb2 = nbase + hi * 16;
        v16bf bv;
        #pragma unroll
        for (int e = 0; e < 16; ++e) bv[e] = shg[(nb2 + e) * 16 + col];
        acc = __builtin_amdgcn_wmma_f32_16x16x32_bf16(false, av, false, bv, (short)0, acc, false, false);
    }
    int k = lane & 15;
    int obase = w * 16 + hi * 8;
    if (k < K) {
        #pragma unroll
        for (int r = 0; r < 8; ++r)
            y[(size_t)b * KC + (size_t)(obase + r) * K + k] = acc[r];
    }
}

// ============================================================
// K6b: exact train-mode BN over (B,K) + ReLU; one block per channel o.
// ============================================================
__global__ void k_bnrelu(const float* __restrict__ y, const float* __restrict__ gamma,
                         const float* __restrict__ beta, float* __restrict__ out) {
    int o = blockIdx.x, t = threadIdx.x, b = t >> 3, k = t & 7;
    float v = y[(size_t)b * KC + o * K + k];
    __shared__ float red[128], red2[128];
    red[t] = v; red2[t] = v * v;
    __syncthreads();
    for (int s = 64; s > 0; s >>= 1) {
        if (t < s) { red[t] += red[t + s]; red2[t] += red2[t + s]; }
        __syncthreads();
    }
    float mean = red[0] * (1.f / 128.f);
    float var  = red2[0] * (1.f / 128.f) - mean * mean;
    float sc   = gamma[o] * rsqrtf(var + EPS_BN);
    out[(size_t)b * KC + o * K + k] = fmaxf(sc * (v - mean) + beta[o], 0.f);
}

// ============================================================
// K7: mutual interaction (8x8 softmax, 2-step cascade GCN, fold cw) -> Mfin[b][C][K]
// ============================================================
__global__ void k_mutual(const float* __restrict__ edge, const float* __restrict__ r1,
                         const float* __restrict__ r2, const float* __restrict__ W1,
                         const float* __restrict__ W2, const float* __restrict__ cw,
                         float* __restrict__ Mfin) {
    int b = blockIdx.x, t = threadIdx.x;
    __shared__ float a_sh[K * K];
    __shared__ float adj_sh[K * K];
    __shared__ float x_sh[C * K];
    const float* eb  = edge + (size_t)b * KC;
    const float* r1b = r1   + (size_t)b * KC;
    const float* r2b = r2   + (size_t)b * KC;

    if (t < K * K) {
        int k = t >> 3, l = t & 7;
        float acc = 0.f;
        for (int i = 0; i < C; ++i) acc += eb[i * K + k] * r1b[i * K + l];
        a_sh[t] = acc;
    }
    __syncthreads();
    if (t < K) {
        float mx = -1e30f;
        for (int l = 0; l < K; ++l) mx = fmaxf(mx, a_sh[t * K + l]);
        float sm = 0.f;
        for (int l = 0; l < K; ++l) { float e = __expf(a_sh[t * K + l] - mx); a_sh[t * K + l] = e; sm += e; }
        float inv = 1.f / sm;
        for (int l = 0; l < K; ++l) a_sh[t * K + l] *= inv;
    }
    __syncthreads();
    {
        float rr[K], ee[K];
        #pragma unroll
        for (int l = 0; l < K; ++l) rr[l] = r2b[t * K + l];
        #pragma unroll
        for (int k = 0; k < K; ++k) ee[k] = eb[t * K + k];
        #pragma unroll
        for (int k = 0; k < K; ++k) {
            float mm = 0.f;
            #pragma unroll
            for (int l = 0; l < K; ++l) mm += a_sh[k * K + l] * rr[l];
            x_sh[t * K + k] = ee[k] + mm;
        }
    }
    __syncthreads();
    for (int it = 0; it < 2; ++it) {
        const float* Wg = (it == 0) ? W1 : W2;
        if (t < K * K) {
            int k = t >> 3, l = t & 7;
            float acc = 0.f;
            for (int i = 0; i < C; ++i) acc += x_sh[i * K + k] * x_sh[i * K + l];
            adj_sh[t] = acc;
        }
        __syncthreads();
        if (t < K) {
            float mx = -1e30f;
            for (int l = 0; l < K; ++l) mx = fmaxf(mx, adj_sh[t * K + l]);
            float sm = 0.f;
            for (int l = 0; l < K; ++l) { float e = __expf(adj_sh[t * K + l] - mx); adj_sh[t * K + l] = e; sm += e; }
            float inv = 1.f / sm;
            for (int l = 0; l < K; ++l) adj_sh[t * K + l] *= inv;
        }
        __syncthreads();
        float sup[K];
        #pragma unroll
        for (int l = 0; l < K; ++l) sup[l] = 0.f;
        for (int i = 0; i < C; ++i) {
            float wv = Wg[(size_t)i * C + t];
            #pragma unroll
            for (int l = 0; l < K; ++l) sup[l] += x_sh[i * K + l] * wv;
        }
        float xn[K];
        #pragma unroll
        for (int k = 0; k < K; ++k) {
            float acc = 0.f;
            #pragma unroll
            for (int l = 0; l < K; ++l) acc += adj_sh[k * K + l] * sup[l];
            xn[k] = acc;
        }
        __syncthreads();
        #pragma unroll
        for (int k = 0; k < K; ++k) x_sh[t * K + k] = xn[k];
        __syncthreads();
    }
    #pragma unroll
    for (int k = 0; k < K; ++k) x_sh[t * K + k] = fmaxf(x_sh[t * K + k], 0.f);
    __syncthreads();
    {
        float acc[K];
        #pragma unroll
        for (int k = 0; k < K; ++k) acc[k] = 0.f;
        for (int i = 0; i < C; ++i) {
            float wv = cw[(size_t)t * C + i];
            #pragma unroll
            for (int k = 0; k < K; ++k) acc[k] += wv * x_sh[i * K + k];
        }
        #pragma unroll
        for (int k = 0; k < K; ++k) Mfin[(size_t)b * KC + t * K + k] = acc[k];
    }
}

// ============================================================
// K8: exact BN affine for the pixel-space conv from tiny moments
// ============================================================
__global__ void k_stats(const float* __restrict__ Mfin, const float* __restrict__ sred,
                        const float* __restrict__ Gred, const float* __restrict__ gamma,
                        const float* __restrict__ beta, float* __restrict__ Asc, float* __restrict__ Bsh) {
    int o = threadIdx.x;
    float accm = 0.f, acc2 = 0.f;
    for (int b = 0; b < B; ++b) {
        float Mr[K];
        #pragma unroll
        for (int k = 0; k < K; ++k) Mr[k] = Mfin[(size_t)b * KC + o * K + k];
        const float* sb = sred + b * K;
        const float* Gb = Gred + b * K * K;
        #pragma unroll
        for (int k = 0; k < K; ++k) {
            accm += Mr[k] * sb[k];
            float gv = 0.f;
            #pragma unroll
            for (int j = 0; j < K; ++j) gv += Gb[k * K + j] * Mr[j];
            acc2 += Mr[k] * gv;
        }
    }
    const float invBN = 1.f / (float)(B * N);
    float mean = accm * invBN;
    float var  = acc2 * invBN - mean * mean;
    float A    = gamma[o] * rsqrtf(var + EPS_BN);
    Asc[o] = A;
    Bsh[o] = beta[o] - A * mean;
}

// ============================================================
// K9: final streaming pass (4 pixels/thread, b128)
// ============================================================
__global__ void k_final(const float* __restrict__ xin, const float* __restrict__ gate,
                        const float* __restrict__ assign, const float* __restrict__ Mfin,
                        const float* __restrict__ Asc, const float* __restrict__ Bsh,
                        float* __restrict__ out) {
    int bc = blockIdx.x;
    int b  = bc >> 8;
    int c  = bc & 255;
    __shared__ float Mr[K];
    if (threadIdx.x < K) Mr[threadIdx.x] = Mfin[(size_t)b * KC + c * K + threadIdx.x];
    __syncthreads();
    float A = Asc[c], Bb = Bsh[c];
    const float* ab = assign + (size_t)b * KN;
    const float* xr = xin + (size_t)b * CN + (size_t)c * N;
    const float* gb = gate + (size_t)b * N;
    float* ob = out + (size_t)b * CN + (size_t)c * N;
    float m0 = Mr[0], m1 = Mr[1], m2 = Mr[2], m3 = Mr[3];
    float m4 = Mr[4], m5 = Mr[5], m6 = Mr[6], m7 = Mr[7];
    __builtin_prefetch(xr + threadIdx.x * 4, 0, 1);     // global_prefetch
    __builtin_prefetch(ab + threadIdx.x * 4, 0, 1);
    for (int base = threadIdx.x * 4; base < N; base += 1024) {
        float4 y = {0.f, 0.f, 0.f, 0.f};
        y = f4fma(m0, ld4(ab + base),          y);
        y = f4fma(m1, ld4(ab + N + base),      y);
        y = f4fma(m2, ld4(ab + 2 * N + base),  y);
        y = f4fma(m3, ld4(ab + 3 * N + base),  y);
        y = f4fma(m4, ld4(ab + 4 * N + base),  y);
        y = f4fma(m5, ld4(ab + 5 * N + base),  y);
        y = f4fma(m6, ld4(ab + 6 * N + base),  y);
        y = f4fma(m7, ld4(ab + 7 * N + base),  y);
        float4 x4 = ld4(xr + base);
        float4 g4 = ld4(gb + base);
        float4 o4;
        o4.x = g4.x * x4.x + fmaxf(A * y.x + Bb, 0.f);
        o4.y = g4.y * x4.y + fmaxf(A * y.y + Bb, 0.f);
        o4.z = g4.z * x4.z + fmaxf(A * y.z + Bb, 0.f);
        o4.w = g4.w * x4.w + fmaxf(A * y.w + Bb, 0.f);
        *(float4*)(ob + base) = o4;
    }
}

// ============================================================
// host-side orchestration
// ============================================================
extern "C" void kernel_launch(void* const* d_in, const int* in_sizes, int n_in,
                              void* d_out, int out_size, void* d_ws, size_t ws_size,
                              hipStream_t stream) {
    (void)in_sizes; (void)n_in; (void)out_size; (void)ws_size;

    const float* rgb        = (const float*)d_in[0];
    const float* t          = (const float*)d_in[1];
    const float* edger      = (const float*)d_in[2];
    const float* edget      = (const float*)d_in[3];
    const float* pred_w     = (const float*)d_in[4];
    const float* pred_b     = (const float*)d_in[5];
    const float* rgb_anchor = (const float*)d_in[6];
    const float* rgb_sigma  = (const float*)d_in[7];
    const float* t_anchor   = (const float*)d_in[8];
    const float* t_sigma    = (const float*)d_in[9];
    const float* rgb_c1_w = (const float*)d_in[10]; const float* rgb_c1_g = (const float*)d_in[11]; const float* rgb_c1_b = (const float*)d_in[12];
    const float* rgb_c2_w = (const float*)d_in[13]; const float* rgb_c2_g = (const float*)d_in[14]; const float* rgb_c2_b = (const float*)d_in[15];
    const float* t_c1_w   = (const float*)d_in[16]; const float* t_c1_g   = (const float*)d_in[17]; const float* t_c1_b   = (const float*)d_in[18];
    const float* t_c2_w   = (const float*)d_in[19]; const float* t_c2_g   = (const float*)d_in[20]; const float* t_c2_b   = (const float*)d_in[21];
    const float* t2r_w    = (const float*)d_in[22]; const float* t2r_g    = (const float*)d_in[23]; const float* t2r_b    = (const float*)d_in[24];
    const float* r2t_w    = (const float*)d_in[25]; const float* r2t_g    = (const float*)d_in[26]; const float* r2t_b    = (const float*)d_in[27];
    const float* t2r_g1_w = (const float*)d_in[28]; const float* t2r_g2_w = (const float*)d_in[29];
    const float* r2t_g1_w = (const float*)d_in[30]; const float* r2t_g2_w = (const float*)d_in[31];

    float* ws  = (float*)d_ws;
    float* out = (float*)d_out;

    // workspace layout (floats) — ~7.0 MB
    constexpr size_t GATE_R = 0;
    constexpr size_t GATE_T = GATE_R + (size_t)B * N;
    constexpr size_t TABA_R = GATE_T + (size_t)B * N;
    constexpr size_t TABB_R = TABA_R + KC;
    constexpr size_t CST_R  = TABB_R + KC;
    constexpr size_t TABA_T = CST_R + 64;
    constexpr size_t TABB_T = TABA_T + KC;
    constexpr size_t CST_T  = TABB_T + KC;
    constexpr size_t ASG_R  = CST_T + 64;
    constexpr size_t ASG_T  = ASG_R + (size_t)B * KN;
    constexpr size_t S_R    = ASG_T + (size_t)B * KN;
    constexpr size_t S_T    = S_R + B * K;
    constexpr size_t G_R    = S_T + B * K;
    constexpr size_t G_T    = G_R + B * K * K;
    constexpr size_t WXP_R  = G_T + B * K * K;               // 4 partial segments
    constexpr size_t WXP_T  = WXP_R + (size_t)4 * B * KC;
    constexpr size_t YTMP   = WXP_T + (size_t)4 * B * KC;    // scratch for graph GEMMs
    constexpr size_t GR_R   = YTMP + (size_t)B * KC;
    constexpr size_t GR_T   = GR_R + (size_t)B * KC;
    constexpr size_t G1R    = GR_T + (size_t)B * KC;
    constexpr size_t G2R    = G1R + (size_t)B * KC;
    constexpr size_t G1T    = G2R + (size_t)B * KC;
    constexpr size_t G2T    = G1T + (size_t)B * KC;
    constexpr size_t MF_R   = G2T + (size_t)B * KC;
    constexpr size_t MF_T   = MF_R + (size_t)B * KC;
    constexpr size_t A_R    = MF_T + (size_t)B * KC;
    constexpr size_t B_R    = A_R + C;
    constexpr size_t A_T    = B_R + C;
    constexpr size_t B_T    = A_T + C;

    // 1. edge gates
    k_gate<<<B * N / 1024, 256, 0, stream>>>(edger, edget, pred_w, pred_b, ws + GATE_R, ws + GATE_T);
    // 2. cluster tables + zero moment accumulators
    k_tables<<<K, 256, 0, stream>>>(rgb_anchor, rgb_sigma, ws + TABA_R, ws + TABB_R, ws + CST_R);
    k_tables<<<K, 256, 0, stream>>>(t_anchor, t_sigma, ws + TABA_T, ws + TABB_T, ws + CST_T);
    k_zero<<<(2 * B * K + 2 * B * K * K + 255) / 256, 256, 0, stream>>>(ws + S_R, 2 * B * K + 2 * B * K * K);
    // 3. soft assignment + moments
    k_assign<<<B * N / 1024, 256, 0, stream>>>(rgb, ws + GATE_R, ws + TABA_R, ws + TABB_R, ws + CST_R,
                                               ws + ASG_R, ws + S_R, ws + G_R);
    k_assign<<<B * N / 1024, 256, 0, stream>>>(t, ws + GATE_T, ws + TABA_T, ws + TABB_T, ws + CST_T,
                                               ws + ASG_T, ws + S_T, ws + G_T);
    // 4. WMMA: wx = assign @ xf (4 partial segments each)
    k_wx<<<B * 4, 256, 0, stream>>>(rgb, ws + GATE_R, ws + ASG_R, ws + WXP_R);
    k_wx<<<B * 4, 256, 0, stream>>>(t, ws + GATE_T, ws + ASG_T, ws + WXP_T);
    // 5. nodes + L2 norms -> graphs
    k_nodes<<<B, 256, 0, stream>>>(ws + WXP_R, ws + S_R, rgb_anchor, rgb_sigma, ws + GR_R);
    k_nodes<<<B, 256, 0, stream>>>(ws + WXP_T, ws + S_T, t_anchor, t_sigma, ws + GR_T);
    // 6. graph convs: WMMA GEMM + exact BN/ReLU
    k_ggemm<<<B, 512, 0, stream>>>(ws + GR_R, rgb_c1_w, ws + YTMP);
    k_bnrelu<<<C, 128, 0, stream>>>(ws + YTMP, rgb_c1_g, rgb_c1_b, ws + G1R);
    k_ggemm<<<B, 512, 0, stream>>>(ws + GR_R, rgb_c2_w, ws + YTMP);
    k_bnrelu<<<C, 128, 0, stream>>>(ws + YTMP, rgb_c2_g, rgb_c2_b, ws + G2R);
    k_ggemm<<<B, 512, 0, stream>>>(ws + GR_T, t_c1_w, ws + YTMP);
    k_bnrelu<<<C, 128, 0, stream>>>(ws + YTMP, t_c1_g, t_c1_b, ws + G1T);
    k_ggemm<<<B, 512, 0, stream>>>(ws + GR_T, t_c2_w, ws + YTMP);
    k_bnrelu<<<C, 128, 0, stream>>>(ws + YTMP, t_c2_g, t_c2_b, ws + G2T);
    // 7. mutual interaction -> folded M matrices
    k_mutual<<<B, 256, 0, stream>>>(ws + GR_R, ws + G1T, ws + G2T, t2r_g1_w, t2r_g2_w, t2r_w, ws + MF_R);
    k_mutual<<<B, 256, 0, stream>>>(ws + GR_T, ws + G1R, ws + G2R, r2t_g1_w, r2t_g2_w, r2t_w, ws + MF_T);
    // 8. exact BN affines from tiny moments
    k_stats<<<1, 256, 0, stream>>>(ws + MF_R, ws + S_R, ws + G_R, t2r_g, t2r_b, ws + A_R, ws + B_R);
    k_stats<<<1, 256, 0, stream>>>(ws + MF_T, ws + S_T, ws + G_T, r2t_g, r2t_b, ws + A_T, ws + B_T);
    // 9. final streaming pass
    k_final<<<B * C, 256, 0, stream>>>(rgb, ws + GATE_R, ws + ASG_R, ws + MF_R, ws + A_R, ws + B_R, out);
    k_final<<<B * C, 256, 0, stream>>>(t, ws + GATE_T, ws + ASG_T, ws + MF_T, ws + A_T, ws + B_T,
                                       out + (size_t)B * CN);
}
